// MoeLayer_89446988906922
// MI455X (gfx1250) — matrix operand order repaired and checked by reading, not compiled, not run on hardware
//
#include <hip/hip_runtime.h>

#define E_ 8
#define K_ 2
#define D_ 1024
#define F_ 2048
#define T_ 2048

typedef __attribute__((ext_vector_type(8)))  float          v8f;
typedef __attribute__((ext_vector_type(16))) __bf16         v16bf;
typedef __attribute__((ext_vector_type(16))) unsigned short v16us;
typedef __attribute__((ext_vector_type(8)))  unsigned short v8us;

// ---- workspace byte offsets ----
#define OFF_CNT   ((size_t)0)                          // E ints
#define OFF_IDX   ((size_t)256)                        // T*K ints
#define OFF_W     (OFF_IDX + (size_t)T_*K_*4)          // T*K floats
#define OFF_LIST  (OFF_W   + (size_t)T_*K_*4)          // E*T ints
#define OFF_XB    ((size_t)0x20000)                    // T*D bf16      (4MB)
#define OFF_HBUF  ((size_t)0x500000)                   // T*K*F bf16    (16MB)
#define OFF_YBUF  (OFF_HBUF + (size_t)T_*K_*F_*2)      // K*T*D f32     (16MB)
#define OFF_W1T   (OFF_YBUF + (size_t)K_*T_*D_*4)      // E*F*D bf16    (32MB)
#define OFF_W2T   (OFF_W1T  + (size_t)E_*F_*D_*2)      // E*D*F bf16    (32MB)

__device__ __forceinline__ unsigned short f2bf(float f) {
    unsigned int u = __float_as_uint(f);
    u += 0x7FFFu + ((u >> 16) & 1u);           // round-to-nearest-even
    return (unsigned short)(u >> 16);
}

__device__ __forceinline__ float silu_fast(float h) {
    // h * sigmoid(h) with hardware v_rcp (bf16 output -> ample precision)
    return h * __builtin_amdgcn_rcpf(1.0f + __expf(-h));
}

__device__ __forceinline__ v16bf mk_a(const unsigned short* row, int kk, int sel) {
    v8us a0 = *reinterpret_cast<const v8us*>(row + kk + sel * 16);
    v8us a1 = *reinterpret_cast<const v8us*>(row + kk + sel * 16 + 8);
    v16us au;
#pragma unroll
    for (int i = 0; i < 8; ++i) { au[i] = a0[i]; au[i + 8] = a1[i]; }
    return __builtin_bit_cast(v16bf, au);
}

// ---------------- pre-pass: x f32 -> bf16 ----------------
__global__ void moe_cvt_x(const float* __restrict__ x, unsigned short* __restrict__ xb) {
    size_t i = ((size_t)blockIdx.x * 256 + threadIdx.x) * 8;
    float4 v0 = *reinterpret_cast<const float4*>(x + i);
    float4 v1 = *reinterpret_cast<const float4*>(x + i + 4);
    ushort4 u0, u1;
    u0.x = f2bf(v0.x); u0.y = f2bf(v0.y); u0.z = f2bf(v0.z); u0.w = f2bf(v0.w);
    u1.x = f2bf(v1.x); u1.y = f2bf(v1.y); u1.z = f2bf(v1.z); u1.w = f2bf(v1.w);
    *reinterpret_cast<ushort4*>(xb + i)     = u0;
    *reinterpret_cast<ushort4*>(xb + i + 4) = u1;
}

// ---------------- pre-pass: per-expert transpose+convert  [R][C] f32 -> [C][R] bf16 ----
__global__ void moe_cvt_wt(const float* __restrict__ src, unsigned short* __restrict__ dst,
                           int R, int C) {
    __shared__ unsigned short tile[32][33];
    int e = blockIdx.z;
    size_t base = (size_t)e * R * C;
    int c0 = blockIdx.x * 32, r0 = blockIdx.y * 32;
    int tx = threadIdx.x, ty = threadIdx.y;        // blockDim (32, 8)
#pragma unroll
    for (int i = 0; i < 32; i += 8)
        tile[ty + i][tx] = f2bf(src[base + (size_t)(r0 + ty + i) * C + (c0 + tx)]);
    __syncthreads();
#pragma unroll
    for (int i = 0; i < 32; i += 8)
        dst[base + (size_t)(c0 + ty + i) * R + (r0 + tx)] = tile[tx][ty + i];
}

// ---------------- reset counters ----------------
__global__ void moe_init(int* counts) {
    if (threadIdx.x < E_) counts[threadIdx.x] = 0;
}

// ---------------- gate + top-2 + routing lists ----------------
__global__ void __launch_bounds__(128) moe_gate(
        const float* __restrict__ x, const float* __restrict__ Wg,
        const float* __restrict__ bg, int* __restrict__ counts,
        int* __restrict__ topk_idx, float* __restrict__ topk_w,
        int* __restrict__ lists) {
    int wave = threadIdx.x >> 5, lane = threadIdx.x & 31;
    int t = blockIdx.x * 4 + wave;
    const float* xr = x + (size_t)t * D_;
    float acc[E_] = {};
    for (int d = lane; d < D_; d += 32) {
        float xv = xr[d];
        const float4* wg4 = reinterpret_cast<const float4*>(Wg + (size_t)d * E_);
        float4 a = wg4[0], b = wg4[1];
        acc[0] += xv * a.x; acc[1] += xv * a.y; acc[2] += xv * a.z; acc[3] += xv * a.w;
        acc[4] += xv * b.x; acc[5] += xv * b.y; acc[6] += xv * b.z; acc[7] += xv * b.w;
    }
#pragma unroll
    for (int e = 0; e < E_; ++e)
#pragma unroll
        for (int off = 16; off; off >>= 1)
            acc[e] += __shfl_xor(acc[e], off, 32);
    if (lane == 0) {
        float lg[E_];
#pragma unroll
        for (int e = 0; e < E_; ++e) lg[e] = acc[e] + bg[e];
        int i0 = 0;
#pragma unroll
        for (int e = 1; e < E_; ++e) if (lg[e] > lg[i0]) i0 = e;   // first max on ties
        int i1 = (i0 == 0) ? 1 : 0;
#pragma unroll
        for (int e = 0; e < E_; ++e) if (e != i0 && lg[e] > lg[i1]) i1 = e;
        float e1 = expf(lg[i1] - lg[i0]);       // lg[i0] >= lg[i1]
        float s  = 1.0f + e1;
        topk_idx[t*2 + 0] = i0;  topk_idx[t*2 + 1] = i1;
        topk_w  [t*2 + 0] = 1.0f / s;
        topk_w  [t*2 + 1] = e1 / s;
        int p0 = atomicAdd(counts + i0, 1); lists[i0*T_ + p0] = t;               // slot 0
        int p1 = atomicAdd(counts + i1, 1); lists[i1*T_ + p1] = t | (1 << 16);   // slot 1
    }
}

// ---------------- h = silu(x @ W1[e]) : M=32 x N=128 tiles, direct-fragment WMMA ----
__global__ void __launch_bounds__(128) moe_ffn1(
        const unsigned short* __restrict__ xb, const unsigned short* __restrict__ W1t,
        const int* __restrict__ counts, const int* __restrict__ lists,
        unsigned short* __restrict__ hbuf) {
    int e = blockIdx.z, mt = blockIdx.y;
    int cnt = counts[e];
    if (mt * 32 >= cnt) return;
    int n0 = blockIdx.x * 128;

    __shared__ int rowTok[32];
    __shared__ int rowPair[32];
    int tid = threadIdx.x;
    if (tid < 32) {
        int li = mt * 32 + tid; if (li > cnt - 1) li = cnt - 1;   // pad = dup last (benign)
        int en = lists[e * T_ + li];
        rowTok[tid] = en & 0xFFFF;
        rowPair[tid] = (en & 0xFFFF) * 2 + (en >> 16);
    }
    __syncthreads();

    int lane = tid & 31, wave = tid >> 5, sel = lane >> 4, l15 = lane & 15;
    int col0 = n0 + wave * 32 + l15, col1 = col0 + 16;
    const unsigned short* ar0 = xb + (size_t)rowTok[l15] * D_;        // A rows 0..15
    const unsigned short* ar1 = xb + (size_t)rowTok[l15 + 16] * D_;   // A rows 16..31
    const unsigned short* bc0 = W1t + ((size_t)e * F_ + col0) * D_;   // B col, K-contig
    const unsigned short* bc1 = W1t + ((size_t)e * F_ + col1) * D_;

    v8f acc00 = {}, acc01 = {}, acc10 = {}, acc11 = {};
    for (int kk = 0; kk < D_; kk += 32) {
        __builtin_prefetch(bc0 + kk + 512, 0, 1);
        __builtin_prefetch(bc1 + kk + 512, 0, 1);
        v16bf av0 = mk_a(ar0, kk, sel);
        v16bf av1 = mk_a(ar1, kk, sel);
        v16bf bv0 = __builtin_bit_cast(v16bf,
                        *reinterpret_cast<const v16us*>(bc0 + kk + sel * 16));
        v16bf bv1 = __builtin_bit_cast(v16bf,
                        *reinterpret_cast<const v16us*>(bc1 + kk + sel * 16));
        acc00 = __builtin_amdgcn_wmma_f32_16x16x32_bf16(false, av0, false, bv0, (short)0, acc00, false, false);
        acc01 = __builtin_amdgcn_wmma_f32_16x16x32_bf16(false, av0, false, bv1, (short)0, acc01, false, false);
        acc10 = __builtin_amdgcn_wmma_f32_16x16x32_bf16(false, av1, false, bv0, (short)0, acc10, false, false);
        acc11 = __builtin_amdgcn_wmma_f32_16x16x32_bf16(false, av1, false, bv1, (short)0, acc11, false, false);
    }
#pragma unroll
    for (int v = 0; v < 8; ++v) {
        int m0 = v + sel * 8, m1 = m0 + 16;
        size_t b0 = (size_t)rowPair[m0] * F_;
        size_t b1 = (size_t)rowPair[m1] * F_;
        hbuf[b0 + col0] = f2bf(silu_fast(acc00[v]));
        hbuf[b0 + col1] = f2bf(silu_fast(acc01[v]));
        hbuf[b1 + col0] = f2bf(silu_fast(acc10[v]));
        hbuf[b1 + col1] = f2bf(silu_fast(acc11[v]));
    }
}

// ---------------- y = (h @ W2[e]) * w -> ybuf[slot][token] ----------------
__global__ void __launch_bounds__(128) moe_ffn2(
        const unsigned short* __restrict__ hbuf, const unsigned short* __restrict__ W2t,
        const int* __restrict__ counts, const int* __restrict__ lists,
        const float* __restrict__ topk_w, float* __restrict__ ybuf) {
    int e = blockIdx.z, mt = blockIdx.y;
    int cnt = counts[e];
    if (mt * 32 >= cnt) return;
    int n0 = blockIdx.x * 128;

    __shared__ int   rowTok[32], rowSlot[32], rowPair[32];
    __shared__ float rowW[32];
    int tid = threadIdx.x;
    if (tid < 32) {
        int li = mt * 32 + tid; if (li > cnt - 1) li = cnt - 1;
        int en = lists[e * T_ + li];
        int tok = en & 0xFFFF, slot = en >> 16;
        rowTok[tid] = tok; rowSlot[tid] = slot; rowPair[tid] = tok * 2 + slot;
        rowW[tid] = topk_w[tok * 2 + slot];
    }
    __syncthreads();

    int lane = tid & 31, wave = tid >> 5, sel = lane >> 4, l15 = lane & 15;
    int col0 = n0 + wave * 32 + l15, col1 = col0 + 16;
    const unsigned short* ar0 = hbuf + (size_t)rowPair[l15] * F_;
    const unsigned short* ar1 = hbuf + (size_t)rowPair[l15 + 16] * F_;
    const unsigned short* bc0 = W2t + ((size_t)e * D_ + col0) * F_;   // K(=f)-contig
    const unsigned short* bc1 = W2t + ((size_t)e * D_ + col1) * F_;

    v8f acc00 = {}, acc01 = {}, acc10 = {}, acc11 = {};
    for (int kk = 0; kk < F_; kk += 32) {
        __builtin_prefetch(bc0 + kk + 512, 0, 1);
        __builtin_prefetch(bc1 + kk + 512, 0, 1);
        v16bf av0 = mk_a(ar0, kk, sel);
        v16bf av1 = mk_a(ar1, kk, sel);
        v16bf bv0 = __builtin_bit_cast(v16bf,
                        *reinterpret_cast<const v16us*>(bc0 + kk + sel * 16));
        v16bf bv1 = __builtin_bit_cast(v16bf,
                        *reinterpret_cast<const v16us*>(bc1 + kk + sel * 16));
        acc00 = __builtin_amdgcn_wmma_f32_16x16x32_bf16(false, av0, false, bv0, (short)0, acc00, false, false);
        acc01 = __builtin_amdgcn_wmma_f32_16x16x32_bf16(false, av0, false, bv1, (short)0, acc01, false, false);
        acc10 = __builtin_amdgcn_wmma_f32_16x16x32_bf16(false, av1, false, bv0, (short)0, acc10, false, false);
        acc11 = __builtin_amdgcn_wmma_f32_16x16x32_bf16(false, av1, false, bv1, (short)0, acc11, false, false);
    }
#pragma unroll
    for (int v = 0; v < 8; ++v) {
        int m0 = v + sel * 8, m1 = m0 + 16;
        size_t o0 = ((size_t)rowSlot[m0] * T_ + rowTok[m0]) * D_;
        size_t o1 = ((size_t)rowSlot[m1] * T_ + rowTok[m1]) * D_;
        float w0 = rowW[m0], w1 = rowW[m1];
        ybuf[o0 + col0] = acc00[v] * w0;
        ybuf[o0 + col1] = acc01[v] * w0;
        ybuf[o1 + col0] = acc10[v] * w1;
        ybuf[o1 + col1] = acc11[v] * w1;
    }
}

// ---------------- out = ybuf0 + ybuf1 ----------------
__global__ void moe_combine(const float* __restrict__ ybuf, float* __restrict__ out) {
    size_t i = (size_t)blockIdx.x * blockDim.x + threadIdx.x;     // float4 index
    const float4* y0 = reinterpret_cast<const float4*>(ybuf);
    const float4* y1 = reinterpret_cast<const float4*>(ybuf + (size_t)T_ * D_);
    float4 a = y0[i], b = y1[i];
    float4 r; r.x = a.x + b.x; r.y = a.y + b.y; r.z = a.z + b.z; r.w = a.w + b.w;
    reinterpret_cast<float4*>(out)[i] = r;
}

extern "C" void kernel_launch(void* const* d_in, const int* in_sizes, int n_in,
                              void* d_out, int out_size, void* d_ws, size_t ws_size,
                              hipStream_t stream) {
    const float* x  = (const float*)d_in[0];
    const float* Wg = (const float*)d_in[1];
    const float* bg = (const float*)d_in[2];
    const float* W1 = (const float*)d_in[3];
    const float* W2 = (const float*)d_in[4];

    char* ws = (char*)d_ws;
    int*   counts = (int*)  (ws + OFF_CNT);
    int*   tidx   = (int*)  (ws + OFF_IDX);
    float* tw     = (float*)(ws + OFF_W);
    int*   lists  = (int*)  (ws + OFF_LIST);
    unsigned short* xb   = (unsigned short*)(ws + OFF_XB);
    unsigned short* hbuf = (unsigned short*)(ws + OFF_HBUF);
    float*          ybuf = (float*)         (ws + OFF_YBUF);
    unsigned short* w1t  = (unsigned short*)(ws + OFF_W1T);
    unsigned short* w2t  = (unsigned short*)(ws + OFF_W2T);
    float* out = (float*)d_out;

    // pre-passes: bf16 conversion + weight transposes (K-contiguous B layout)
    moe_cvt_x<<<(T_ * D_) / (256 * 8), 256, 0, stream>>>(x, xb);
    dim3 tb(32, 8);
    dim3 g1t(F_ / 32, D_ / 32, E_);
    moe_cvt_wt<<<g1t, tb, 0, stream>>>(W1, w1t, D_, F_);   // [D][F] -> [F][D]
    dim3 g2t(D_ / 32, F_ / 32, E_);
    moe_cvt_wt<<<g2t, tb, 0, stream>>>(W2, w2t, F_, D_);   // [F][D] -> [D][F]

    moe_init<<<1, 32, 0, stream>>>(counts);
    moe_gate<<<T_ / 4, 128, 0, stream>>>(x, Wg, bg, counts, tidx, tw, lists);

    dim3 g1(F_ / 128, T_ / 32, E_);
    moe_ffn1<<<g1, 128, 0, stream>>>(xb, w1t, counts, lists, hbuf);
    dim3 g2(D_ / 128, T_ / 32, E_);
    moe_ffn2<<<g2, 128, 0, stream>>>(hbuf, w2t, counts, lists, tw, ybuf);
    moe_combine<<<(T_ * D_ / 4) / 256, 256, 0, stream>>>(ybuf, out);
    (void)in_sizes; (void)n_in; (void)out_size; (void)ws_size;
}
